// SocialCircle_29798483100398
// MI455X (gfx1250) — compile-verified, matrix operand order checked by reading, not run on hardware
//
#include <hip/hip_runtime.h>
#include <hip/hip_bf16.h>
#include <math.h>

// ---------------------------------------------------------------------------
// SocialCircle for MI455X (gfx1250, wave32, WMMA).
// All dense matmuls (GRU recurrences, sector MLP, fusion MLP) are mapped to
// v_wmma_f32_16x16x32_f16: f16 operands, f32 accumulation.
// ---------------------------------------------------------------------------

typedef __attribute__((ext_vector_type(16))) _Float16 v16h;
typedef __attribute__((ext_vector_type(8)))  float    v8f;

#define B_SZ   2048
#define N_NBR  64
#define HID    128
#define HN     64
#define SEC    8
#define FEAT_  64
#define COMB_K 1152   // 128 (target feat) + 8*128 (sector enc)

union FragU { v16h v; uint32_t u[8]; };

// Load a 16x32 f16 WMMA operand fragment from a row-major f16 matrix
// (leading dim `ld`, base row `row0`, K offset `k0`).
// Layout per CDNA5 ISA 7.12.2 (16-bit A 16x32), lane L: grp=L>>4, r=L&15;
//   VGPR v in 0..3 -> K = 8*grp + 2v + {0,1}
//   VGPR v in 4..7 -> K = 16 + 8*grp + 2(v-4) + {0,1}
// B (32x16, row-major [N][K] source) uses the symmetric mapping with row0 = n0.
__device__ __forceinline__ v16h load_frag16(const _Float16* base, int row0, int ld, int k0) {
  int lane = threadIdx.x & 31;
  int grp  = lane >> 4;
  int r    = lane & 15;
  FragU f;
  const _Float16* p = base + (size_t)(row0 + r) * ld + k0 + 8 * grp;
#pragma unroll
  for (int v = 0; v < 8; ++v) {
    int ko = (v < 4) ? (2 * v) : (8 + 2 * v);   // pairs of consecutive K
    uint32_t t;
    __builtin_memcpy(&t, p + ko, 4);
    f.u[v] = t;
  }
  return f.v;
}

__device__ __forceinline__ v8f wmma_f16(v16h a, v16h b, v8f c) {
  return __builtin_amdgcn_wmma_f32_16x16x32_f16(false, a, false, b, (short)0, c, false, false);
}

// ---------------------------------------------------------------------------
// f32 -> f16 weight conversion with optional K padding (zeros).
// ---------------------------------------------------------------------------
__global__ void convert_pad_f16(const float* __restrict__ src, _Float16* __restrict__ dst,
                                int rows, int ksrc, int kdst) {
  long long i = (long long)blockIdx.x * blockDim.x + threadIdx.x;
  long long total = (long long)rows * kdst;
  if (i >= total) return;
  int r = (int)(i / kdst);
  int k = (int)(i - (long long)r * kdst);
  dst[i] = (_Float16)(k < ksrc ? src[(long long)r * ksrc + k] : 0.0f);
}

// ---------------------------------------------------------------------------
// GRU over T=8 steps; each wave owns 16 sequences. h kept in LDS as f16;
// recurrence h @ Whh^T (Whh pre-converted to f16, streamed from L2) via WMMA.
// Gate nonlinearity is a VALU phase over LDS pre-activations.
// ---------------------------------------------------------------------------
template <int H, int NW>
__global__ void __launch_bounds__(NW * 32)
gru_kernel(const float* __restrict__ x,        // [Btot][8][2]
           const float* __restrict__ Wih,      // [3H][2] f32
           const _Float16* __restrict__ WhhH,  // [3H][H] f16
           const float* __restrict__ bih,      // [3H]
           const float* __restrict__ bhh,      // [3H]
           _Float16* __restrict__ outh,        // f16 dest, leading dim ld_out
           int ld_out) {
  constexpr int G = 3 * H;
  constexpr int SEQ = 16 * NW;

  __shared__ float    xs[SEQ * 16];      // all 8 steps of (x0,x1) per sequence
  __shared__ _Float16 hh[SEQ * H];       // current hidden state, f16
  __shared__ float    hp[SEQ * G];       // h @ Whh^T + bhh pre-activations

  const int tid  = threadIdx.x;
  const int wave = tid >> 5;
  const int lane = tid & 31;
  const int grp  = lane >> 4;
  const int cn   = lane & 15;
  const int m0   = wave * 16;
  const size_t seqbase = (size_t)blockIdx.x * SEQ;

  for (int i = tid; i < SEQ * 16; i += NW * 32) xs[i] = x[seqbase * 16 + i];
  for (int i = tid; i < SEQ * H; i += NW * 32) hh[i] = (_Float16)0.0f;
  __syncthreads();

  for (int t = 0; t < 8; ++t) {
    // ---- WMMA phase: hp = h @ Whh^T + bhh ----
    v16h a[H / 32];
#pragma unroll
    for (int kc = 0; kc < H / 32; ++kc) a[kc] = load_frag16(hh, m0, H, kc * 32);

    for (int nt = 0; nt < G / 16; ++nt) {
      v8f c = {};
#pragma unroll
      for (int kc = 0; kc < H / 32; ++kc) {
        v16h b = load_frag16(WhhH, nt * 16, H, kc * 32);
        c = wmma_f16(a[kc], b, c);
      }
      const int col = nt * 16 + cn;
      const float bb = bhh[col];
#pragma unroll
      for (int r = 0; r < 8; ++r)
        hp[(m0 + r + 8 * grp) * G + col] = c[r] + bb;
    }
    __syncthreads();

    // ---- elementwise GRU gate phase ----
    for (int i = tid; i < SEQ * H; i += NW * 32) {
      const int s = i / H;
      const int u = i - s * H;
      const float x0 = xs[s * 16 + t * 2];
      const float x1 = xs[s * 16 + t * 2 + 1];
      const float xr = x0 * Wih[2 * u]             + x1 * Wih[2 * u + 1]             + bih[u];
      const float xz = x0 * Wih[2 * (H + u)]       + x1 * Wih[2 * (H + u) + 1]       + bih[H + u];
      const float xn = x0 * Wih[2 * (2 * H + u)]   + x1 * Wih[2 * (2 * H + u) + 1]   + bih[2 * H + u];
      const float hr = hp[s * G + u];
      const float hz = hp[s * G + H + u];
      const float hn = hp[s * G + 2 * H + u];
      const float rr = 1.0f / (1.0f + expf(-(xr + hr)));
      const float zz = 1.0f / (1.0f + expf(-(xz + hz)));
      const float nn = tanhf(xn + rr * hn);
      const float hc = (float)hh[i];
      hh[i] = (_Float16)((1.0f - zz) * nn + zz * hc);
    }
    __syncthreads();
  }

  for (int i = tid; i < SEQ * H; i += NW * 32) {
    const int s = i / H;
    const int u = i - s * H;
    outh[(seqbase + s) * (size_t)ld_out + u] = hh[i];
  }
}

// ---------------------------------------------------------------------------
// Generic WMMA GEMM: D = relu(A @ W^T + bias). A [M][lda] f16, W [Nout][K] f16.
// mode 0: D[row*ldd + col]
// mode 1: scatter per-sector MLP output into comb: row=(b*8+s),
//         D[b*1152 + 128 + s*128 + col]
// ---------------------------------------------------------------------------
template <int NW>
__global__ void __launch_bounds__(NW * 32)
gemm_bias_relu(const _Float16* __restrict__ A, int lda,
               const _Float16* __restrict__ W,
               const float* __restrict__ bias,
               _Float16* __restrict__ D,
               int K, int Nout, int ldd, int mode) {
  const int tid  = threadIdx.x;
  const int wave = tid >> 5;
  const int lane = tid & 31;
  const int grp  = lane >> 4;
  const int cn   = lane & 15;
  const int m0   = (blockIdx.x * NW + wave) * 16;

  for (int nt = 0; nt < Nout / 16; ++nt) {
    v8f c = {};
    for (int kc = 0; kc < K; kc += 32) {
      v16h a = load_frag16(A, m0, lda, kc);
      v16h b = load_frag16(W, nt * 16, K, kc);
      c = wmma_f16(a, b, c);
    }
    const int col = nt * 16 + cn;
    const float bb = bias[col];
#pragma unroll
    for (int r = 0; r < 8; ++r) {
      const int row = m0 + r + 8 * grp;
      float v = c[r] + bb;
      v = v > 0.0f ? v : 0.0f;
      const _Float16 hv = (_Float16)v;
      if (mode == 0) {
        D[(size_t)row * ldd + col] = hv;
      } else {
        const int b_ = row >> 3, s_ = row & 7;
        D[(size_t)b_ * COMB_K + 128 + s_ * 128 + col] = hv;
      }
    }
  }
}

// ---------------------------------------------------------------------------
// Per-batch sector aggregation (tiny FLOPs -> VALU + LDS float atomics).
// Writes padded f16 sector features [B*8][96]: [count, avg_d, avg_vx, avg_vy,
// wfeat(64), zero-pad(28)].
// ---------------------------------------------------------------------------
__global__ void __launch_bounds__(64)
sector_agg(const float* __restrict__ tt,     // [B][8][2]
           const float* __restrict__ ntraj,  // [B][64][8][2]
           const float* __restrict__ ang,    // [B][64]
           const float* __restrict__ mask,   // [B][64]
           const _Float16* __restrict__ nf,  // [B*64][64]
           _Float16* __restrict__ sf) {      // [B*8][96]
  const int b = blockIdx.x;
  const int n = threadIdx.x;  // 0..63 -> one neighbor

  __shared__ float cnt[SEC], sd[SEC], svx[SEC], svy[SEC], sw[SEC];
  __shared__ float wf[SEC][64];

  if (n < SEC) { cnt[n] = 0.f; sd[n] = 0.f; svx[n] = 0.f; svy[n] = 0.f; sw[n] = 0.f; }
  for (int i = n; i < SEC * 64; i += 64) (&wf[0][0])[i] = 0.f;
  __syncthreads();

  const float tx = tt[b * 16 + 14], ty = tt[b * 16 + 15];
  const int base = (b * N_NBR + n) * 16;
  const float lx = ntraj[base + 14], ly = ntraj[base + 15];
  const float px = ntraj[base + 12], py = ntraj[base + 13];
  const float dx = lx - tx, dy = ly - ty;
  const float dist = sqrtf(dx * dx + dy * dy);
  const float vx = lx - px, vy = ly - py;
  const float a = ang[b * N_NBR + n];
  int sid = (int)(a * 1.27323954473516f);  // angle / (2*pi/8), truncation
  sid = sid < 0 ? 0 : (sid > 7 ? 7 : sid);
  const bool valid = mask[b * N_NBR + n] > 0.0f;
  const float w = expf(-dist * 0.1f);      // exp(-dist / MAX_DIST)

  if (valid) {
    atomicAdd(&cnt[sid], 1.0f);
    atomicAdd(&sd[sid], dist);
    atomicAdd(&svx[sid], vx);
    atomicAdd(&svy[sid], vy);
    atomicAdd(&sw[sid], w);
    const _Float16* nfp = nf + (size_t)(b * N_NBR + n) * 64;
    for (int h = 0; h < 64; ++h) atomicAdd(&wf[sid][h], w * (float)nfp[h]);
  }
  __syncthreads();

  for (int i = n; i < SEC * 96; i += 64) {
    const int s = i / 96, c = i - (i / 96) * 96;
    const float cs = cnt[s];
    const float safe = fmaxf(cs, 1.0f);
    const bool ne = cs > 0.0f;
    float v;
    if      (c == 0) v = cs;
    else if (c == 1) v = ne ? sd[s] / safe : 0.0f;
    else if (c == 2) v = ne ? svx[s] / safe : 0.0f;
    else if (c == 3) v = ne ? svy[s] / safe : 0.0f;
    else if (c < 68) v = wf[s][c - 4] / (sw[s] + 1e-8f);
    else             v = 0.0f;
    sf[(size_t)(b * SEC + s) * 96 + c] = (_Float16)v;
  }
}

// ---------------------------------------------------------------------------
// LayerNorm over the last 64 features -> f32 output.
// ---------------------------------------------------------------------------
__global__ void __launch_bounds__(64)
layernorm_out(const _Float16* __restrict__ f, const float* __restrict__ g,
              const float* __restrict__ be, float* __restrict__ out) {
  const int b = blockIdx.x, c = threadIdx.x;
  __shared__ float red[64];
  const float v = (float)f[(size_t)b * 64 + c];
  red[c] = v;
  __syncthreads();
  for (int off = 32; off > 0; off >>= 1) {
    if (c < off) red[c] += red[c + off];
    __syncthreads();
  }
  const float mu = red[0] * (1.0f / 64.0f);
  __syncthreads();
  const float d = v - mu;
  red[c] = d * d;
  __syncthreads();
  for (int off = 32; off > 0; off >>= 1) {
    if (c < off) red[c] += red[c + off];
    __syncthreads();
  }
  const float var = red[0] * (1.0f / 64.0f);
  out[(size_t)b * 64 + c] = d * rsqrtf(var + 1e-5f) * g[c] + be[c];
}

// ---------------------------------------------------------------------------
extern "C" void kernel_launch(void* const* d_in, const int* in_sizes, int n_in,
                              void* d_out, int out_size, void* d_ws, size_t ws_size,
                              hipStream_t stream) {
  (void)in_sizes; (void)n_in; (void)out_size; (void)ws_size;

  const float* tt    = (const float*)d_in[0];   // target_trajectory [2048][8][2]
  const float* ntraj = (const float*)d_in[1];   // neighbor_trajectories [2048][64][8][2]
  const float* ang   = (const float*)d_in[2];
  const float* mask  = (const float*)d_in[3];
  const float* Wih_t = (const float*)d_in[4];
  const float* Whh_t = (const float*)d_in[5];
  const float* bih_t = (const float*)d_in[6];
  const float* bhh_t = (const float*)d_in[7];
  const float* Wih_n = (const float*)d_in[8];
  const float* Whh_n = (const float*)d_in[9];
  const float* bih_n = (const float*)d_in[10];
  const float* bhh_n = (const float*)d_in[11];
  const float* W1  = (const float*)d_in[12];
  const float* b1  = (const float*)d_in[13];
  const float* W2  = (const float*)d_in[14];
  const float* b2  = (const float*)d_in[15];
  const float* Wf1 = (const float*)d_in[16];
  const float* bf1 = (const float*)d_in[17];
  const float* Wf2 = (const float*)d_in[18];
  const float* bf2 = (const float*)d_in[19];
  const float* Wf3 = (const float*)d_in[20];
  const float* bf3 = (const float*)d_in[21];
  const float* lng = (const float*)d_in[22];
  const float* lnb = (const float*)d_in[23];
  float* out = (float*)d_out;

  // ---- workspace carve-up (f16 buffers, 256B aligned) ----
  char* ws = (char*)d_ws;
  size_t off = 0;
  auto carve = [&](size_t bytes) -> char* {
    char* p = ws + off;
    off += (bytes + 255) & ~(size_t)255;
    return p;
  };
  _Float16* whh_n_h = (_Float16*)carve((size_t)192 * 64 * 2);
  _Float16* whh_t_h = (_Float16*)carve((size_t)384 * 128 * 2);
  _Float16* w1_h    = (_Float16*)carve((size_t)128 * 96 * 2);
  _Float16* w2_h    = (_Float16*)carve((size_t)128 * 128 * 2);
  _Float16* wf1_h   = (_Float16*)carve((size_t)256 * COMB_K * 2);
  _Float16* wf2_h   = (_Float16*)carve((size_t)128 * 256 * 2);
  _Float16* wf3_h   = (_Float16*)carve((size_t)64 * 128 * 2);
  _Float16* nf_h    = (_Float16*)carve((size_t)B_SZ * N_NBR * 64 * 2);
  _Float16* comb    = (_Float16*)carve((size_t)B_SZ * COMB_K * 2);
  _Float16* sf      = (_Float16*)carve((size_t)B_SZ * SEC * 96 * 2);
  _Float16* h1      = (_Float16*)carve((size_t)B_SZ * SEC * 128 * 2);
  _Float16* f1b     = (_Float16*)carve((size_t)B_SZ * 256 * 2);
  _Float16* f2b     = (_Float16*)carve((size_t)B_SZ * 128 * 2);
  _Float16* f3b     = (_Float16*)carve((size_t)B_SZ * 64 * 2);

  auto conv = [&](const float* s, _Float16* d, int rows, int ksrc, int kdst) {
    long long total = (long long)rows * kdst;
    int blocks = (int)((total + 255) / 256);
    convert_pad_f16<<<blocks, 256, 0, stream>>>(s, d, rows, ksrc, kdst);
  };
  conv(Whh_n, whh_n_h, 192, 64, 64);
  conv(Whh_t, whh_t_h, 384, 128, 128);
  conv(W1,  w1_h,  128, 68, 96);          // pad K: 68 -> 96
  conv(W2,  w2_h,  128, 128, 128);
  conv(Wf1, wf1_h, 256, COMB_K, COMB_K);
  conv(Wf2, wf2_h, 128, 256, 256);
  conv(Wf3, wf3_h, 64, 128, 128);

  // ---- neighbor GRU: 131072 seqs, H=64, 64 seqs / block (4 waves) ----
  gru_kernel<64, 4><<<(B_SZ * N_NBR) / 64, 128, 0, stream>>>(
      ntraj, Wih_n, whh_n_h, bih_n, bhh_n, nf_h, 64);

  // ---- target GRU: 2048 seqs, H=128, 32 seqs / block (2 waves) ----
  // writes target_feature into comb[b][0..127]
  gru_kernel<128, 2><<<B_SZ / 32, 64, 0, stream>>>(
      tt, Wih_t, whh_t_h, bih_t, bhh_t, comb, COMB_K);

  // ---- sector aggregation ----
  sector_agg<<<B_SZ, 64, 0, stream>>>(tt, ntraj, ang, mask, nf_h, sf);

  // ---- per-sector MLP: (B*S=16384) x 96 -> 128 -> 128 (scatter into comb) ----
  gemm_bias_relu<4><<<(B_SZ * SEC) / 64, 128, 0, stream>>>(sf, 96, w1_h, b1, h1, 96, 128, 128, 0);
  gemm_bias_relu<4><<<(B_SZ * SEC) / 64, 128, 0, stream>>>(h1, 128, w2_h, b2, comb, 128, 128, 0, 1);

  // ---- fusion MLP: 2048 x 1152 -> 256 -> 128 -> 64 ----
  gemm_bias_relu<4><<<B_SZ / 64, 128, 0, stream>>>(comb, COMB_K, wf1_h, bf1, f1b, COMB_K, 256, 256, 0);
  gemm_bias_relu<4><<<B_SZ / 64, 128, 0, stream>>>(f1b, 256, wf2_h, bf2, f2b, 256, 128, 128, 0);
  gemm_bias_relu<4><<<B_SZ / 64, 128, 0, stream>>>(f2b, 128, wf3_h, bf3, f3b, 128, 64, 64, 0);

  // ---- LayerNorm -> f32 out ----
  layernorm_out<<<B_SZ, 64, 0, stream>>>(f3b, lng, lnb, out);
}
